// ResBlock_78408922956471
// MI455X (gfx1250) — matrix operand order repaired and checked by reading, not compiled, not run on hardware
//
#include <hip/hip_runtime.h>

// ---------- types ----------
typedef __attribute__((ext_vector_type(16))) __bf16 v16bf;
typedef __attribute__((ext_vector_type(8)))  __bf16 v8bf;
typedef __attribute__((ext_vector_type(8)))  float  v8f;

#define EPS 1e-12f
#define INV_SQRT2 0.70710678118654752f

// =====================================================================
// Spectral norm: one power iteration. scale_out = 1/sigma.
//   vraw = W^T u ; v = vraw/(|vraw|+eps) ; wv = W v ; sigma = |wv|^2/(|wv|+eps)
// Single block of 256 threads; W is H x K row-major.
// =====================================================================
__global__ void sigma_kernel(const float* __restrict__ W,
                             const float* __restrict__ u,
                             int H, int K,
                             float* __restrict__ vtmp,
                             float* __restrict__ scale_out) {
  __shared__ float red[256];
  __shared__ float sval;
  int tid = threadIdx.x;

  float loc = 0.f;
  for (int k = tid; k < K; k += 256) {
    float s = 0.f;
    for (int h = 0; h < H; ++h) s += W[(long)h * K + k] * u[h];
    vtmp[k] = s;
    loc += s * s;
  }
  red[tid] = loc; __syncthreads();
  for (int off = 128; off > 0; off >>= 1) {
    if (tid < off) red[tid] += red[tid + off];
    __syncthreads();
  }
  if (tid == 0) sval = 1.f / (sqrtf(red[0]) + EPS);
  __syncthreads();
  float inv1 = sval;

  float loc2 = 0.f;
  for (int h = tid; h < H; h += 256) {
    float s = 0.f;
    for (int k = 0; k < K; ++k) s += W[(long)h * K + k] * vtmp[k];
    s *= inv1;
    loc2 += s * s;
  }
  __syncthreads();
  red[tid] = loc2; __syncthreads();
  for (int off = 128; off > 0; off >>= 1) {
    if (tid < off) red[tid] += red[tid + off];
    __syncthreads();
  }
  if (tid == 0) {
    float n2sq = red[0];
    float n2 = sqrtf(n2sq);
    float sigma = n2sq / (n2 + EPS);
    scale_out[0] = 1.f / sigma;
  }
}

// =====================================================================
// Pack OIHW fp32 weight * scale -> bf16 GEMM layout Wp[co][(kh*KW+kw)*Cin+ci]
// =====================================================================
template <int KH, int KW, int CIN>
__global__ void pack_weight(const float* __restrict__ w,
                            const float* __restrict__ scale,
                            __bf16* __restrict__ wp, int Cout) {
  long total = (long)Cout * CIN * KH * KW;
  long i = (long)blockIdx.x * 256 + threadIdx.x;
  if (i >= total) return;
  float s = scale[0];
  long t = i;
  int kw = (int)(t % KW); t /= KW;
  int kh = (int)(t % KH); t /= KH;
  int ci = (int)(t % CIN); t /= CIN;
  int co = (int)t;
  constexpr int KTOT = CIN * KH * KW;
  wp[(long)co * KTOT + (long)(kh * KW + kw) * CIN + ci] = (__bf16)(w[i] * s);
}

// =====================================================================
// fp32 -> bf16 elementwise
// =====================================================================
__global__ void cvt_bf16(const float* __restrict__ in, __bf16* __restrict__ out, long n) {
  long i = (long)blockIdx.x * 256 + threadIdx.x;
  if (i < n) out[i] = (__bf16)in[i];
}

// =====================================================================
// 4x4 FIR blur (upfirdn2d up=1 down=1), kernel outer([1,3,3,1])/64.
// All dims compile-time; out dims = H + 2*PAD - 3. Output bf16.
// =====================================================================
template <typename TIN, int PAD, int HIN, int WIN, int HOUT, int WOUT>
__global__ void blur_kernel(const TIN* __restrict__ in, __bf16* __restrict__ outp,
                            int N, int C) {
  long total = (long)N * C * HOUT * WOUT;
  long i = (long)blockIdx.x * 256 + threadIdx.x;
  if (i >= total) return;
  int wo = (int)(i % WOUT); long t = i / WOUT;
  int ho = (int)(t % HOUT); t /= HOUT;
  long c = t % C;
  long n = t / C;
  const float kv[4] = {0.125f, 0.375f, 0.375f, 0.125f};
  long base = (n * C + c) * (long)(HIN * WIN);
  float acc = 0.f;
#pragma unroll
  for (int iy = 0; iy < 4; ++iy) {
    int hi = ho + iy - PAD;
    if (hi < 0 || hi >= HIN) continue;
#pragma unroll
    for (int ix = 0; ix < 4; ++ix) {
      int wi = wo + ix - PAD;
      if (wi < 0 || wi >= WIN) continue;
      acc += kv[iy] * kv[ix] * (float)in[base + hi * WIN + wi];
    }
  }
  outp[i] = (__bf16)acc;
}

// =====================================================================
// Implicit-GEMM conv via V_WMMA_F32_16X16X32_BF16, all dims compile-time.
//  Block tile: 32 output pixels (consecutive wo in one (n,ho) row) x 128 couts.
//  256 threads = 8 waves: wave = pixSub(2) x coutPair(4); each wave owns TWO
//  16x16 C tiles (couts +0 and +64): each A fragment feeds 2 WMMAs.
//  K staged 128-wide through double-buffered LDS -> 8 WMMAs/wave per barrier.
//  Staging maps lane->pixel so global loads are coalesced along wi.
//  MODE 0: y_bf16 = leaky(acc + b)            (conv1)
//  MODE 1: out_f32 = leaky(acc + b) / sqrt2   (conv2 -> d_out)
//  MODE 2: out_f32 += acc / sqrt2             (skip  -> d_out)
// =====================================================================
template <int MODE, int KH, int KW, int STRIDE, int PAD,
          int CIN, int COUT, int HIN, int WIN, int HOUT, int WOUT>
__global__ void conv_wmma(const __bf16* __restrict__ A,
                          const __bf16* __restrict__ Wp,
                          const float* __restrict__ bias,
                          float* __restrict__ out_f32,
                          __bf16* __restrict__ out_bf) {
  constexpr int KTOT = CIN * KH * KW;
  constexpr int KSTAGE = 128;                   // K per LDS stage
  constexpr int NSTAGE = KTOT / KSTAGE;
  constexpr int CPT = CIN / KSTAGE;             // ci chunks per tap (2)
  constexpr int STRD = KSTAGE + 8;              // 136 halfs = 272B rows (16B-aligned)
  __shared__ __align__(16) __bf16 As[2][32 * STRD];

  constexpr int TPR = WOUT / 32;                // pixel tiles per output row
  int pt = blockIdx.x;
  int wo0 = (pt % TPR) * 32;
  int t2 = pt / TPR;
  int ho = t2 % HOUT;
  int n  = t2 / HOUT;
  int co0 = blockIdx.y * 128;

  int tid = threadIdx.x;
  int lane = tid & 31;
  int wave = tid >> 5;
  int pixSub = wave & 1;     // 16-pixel subtile
  int coutPair = wave >> 1;  // 16-cout subtile pair (+0, +64)

  // fragment lane addressing (ISA rows/cols; k = 16*khalf + s in both A and B)
  int m = pixSub * 16 + (lane & 15);  // A row in block tile
  int khalf = lane >> 4;
  int colA = co0 + coutPair * 16 + (lane & 15);
  int colB = colA + 64;
  const __bf16* wcolA = Wp + (long)colA * KTOT + khalf * 16;
  const __bf16* wcolB = Wp + (long)colB * KTOT + khalf * 16;

  // LDS staging role: lane == pixel (coalesced along wi), 16 channels/thread
  int ldPix = tid & 31;         // lane id -> pixel -> consecutive wi
  int ldCc0 = (tid >> 5) * 16;  // 16 ci values per thread

  auto load_stage = [&](int s, int buf) {
    int tap = s / CPT;                 // constexpr divisor -> shift
    int ci0 = (s - tap * CPT) * KSTAGE;
    int kh = tap / KW;
    int kw = tap - kh * KW;
    int hi = ho * STRIDE + kh - PAD;
    int wi = (wo0 + ldPix) * STRIDE + kw - PAD;
    long chbase = (((long)n * CIN + ci0 + ldCc0) * HIN + hi) * (long)WIN + wi;
    __bf16* dst = &As[buf][ldPix * STRD + ldCc0];
    bool ok = true;
    if constexpr (PAD > 0) ok = (hi >= 0 && hi < HIN && wi >= 0 && wi < WIN);
#pragma unroll
    for (int c = 0; c < 16; ++c) {
      __bf16 val;
      if constexpr (PAD > 0) {
        val = (__bf16)0.0f;
        if (ok) val = A[chbase + (long)c * (HIN * WIN)];
      } else {
        val = A[chbase + (long)c * (HIN * WIN)];  // provably in-bounds when PAD==0
      }
      dst[c] = val;
    }
  };

  v8f acc0 = {};
  v8f acc1 = {};

  auto compute_stage = [&](int s, int buf) {
    const __bf16* as = &As[buf][m * STRD];
#pragma unroll
    for (int j = 0; j < KSTAGE / 32; ++j) {
      // A frag: lane<16 -> K {0..15}; lane>=16 -> K {16..31}
      v8bf alo = *(const v8bf*)&as[j * 32 + khalf * 16];
      v8bf ahi = *(const v8bf*)&as[j * 32 + khalf * 16 + 8];
      v16bf a = __builtin_shufflevector(alo, ahi, 0, 1, 2, 3, 4, 5, 6, 7,
                                        8, 9, 10, 11, 12, 13, 14, 15);
      long kbase = (long)s * KSTAGE + j * 32;
      const __bf16* bpA = wcolA + kbase;
      const __bf16* bpB = wcolB + kbase;
      v16bf b0 = *(const v16bf*)bpA;
      v16bf b1 = *(const v16bf*)bpB;
      __builtin_prefetch(bpA + KSTAGE, 0, 1);   // next stage weights -> GL2
      __builtin_prefetch(bpB + KSTAGE, 0, 1);
      acc0 = __builtin_amdgcn_wmma_f32_16x16x32_bf16(
          false, a, false, b0, (short)0, acc0, false, false);
      acc1 = __builtin_amdgcn_wmma_f32_16x16x32_bf16(
          false, a, false, b1, (short)0, acc1, false, false);
    }
  };

  load_stage(0, 0);
  __syncthreads();
#pragma unroll 2
  for (int s = 0; s < NSTAGE; ++s) {
    int buf = s & 1;
    if (s + 1 < NSTAGE) load_stage(s + 1, buf ^ 1);
    compute_stage(s, buf);
    __syncthreads();
  }

  // epilogue: lanes 0-15 -> N=lane, M=r ; lanes 16-31 -> N=lane-16, M=8+r
  int nl = lane & 15;
  int mh = lane >> 4;
  int cout0 = co0 + coutPair * 16 + nl;
  int cout1 = cout0 + 64;
  float bv0 = 0.f, bv1 = 0.f;
  if constexpr (MODE != 2) {
    bv0 = bias[cout0];
    bv1 = bias[cout1];
  }
#pragma unroll
  for (int r = 0; r < 8; ++r) {
    int mm = pixSub * 16 + mh * 8 + r;
    int wo = wo0 + mm;
    long o0 = (((long)n * COUT + cout0) * HOUT + ho) * (long)WOUT + wo;
    long o1 = (((long)n * COUT + cout1) * HOUT + ho) * (long)WOUT + wo;
    float v0 = acc0[r], v1 = acc1[r];
    if constexpr (MODE == 0) {
      v0 += bv0; v0 = v0 > 0.f ? v0 : 0.2f * v0;
      v1 += bv1; v1 = v1 > 0.f ? v1 : 0.2f * v1;
      out_bf[o0] = (__bf16)v0;
      out_bf[o1] = (__bf16)v1;
    } else if constexpr (MODE == 1) {
      v0 += bv0; v0 = v0 > 0.f ? v0 : 0.2f * v0;
      v1 += bv1; v1 = v1 > 0.f ? v1 : 0.2f * v1;
      out_f32[o0] = v0 * INV_SQRT2;
      out_f32[o1] = v1 * INV_SQRT2;
    } else {
      out_f32[o0] += v0 * INV_SQRT2;
      out_f32[o1] += v1 * INV_SQRT2;
    }
  }
}

// =====================================================================
// Host orchestration
// =====================================================================
extern "C" void kernel_launch(void* const* d_in, const int* in_sizes, int n_in,
                              void* d_out, int out_size, void* d_ws, size_t ws_size,
                              hipStream_t stream) {
  (void)in_sizes; (void)n_in; (void)out_size; (void)ws_size;
  const float* x      = (const float*)d_in[0];
  const float* w1_bar = (const float*)d_in[1];
  const float* u1     = (const float*)d_in[2];
  const float* b1     = (const float*)d_in[4];
  const float* w2_bar = (const float*)d_in[5];
  const float* u2     = (const float*)d_in[6];
  const float* b2     = (const float*)d_in[8];
  const float* ws_bar = (const float*)d_in[9];
  const float* us     = (const float*)d_in[10];
  float* out = (float*)d_out;

  const int Nn = 16, Cin = 256, Cout = 512, H = 64, W = 64;

  char* cur = (char*)d_ws;
  auto alloc = [&](size_t bytes) -> void* {
    void* p = (void*)cur;
    cur += (bytes + 255) & ~(size_t)255;
    return p;
  };
  float*  scales = (float*)alloc(16 * sizeof(float));
  float*  vtmp   = (float*)alloc(2304 * sizeof(float));
  __bf16* xbf    = (__bf16*)alloc((size_t)Nn * Cin * H * W * 2);
  __bf16* w1p    = (__bf16*)alloc((size_t)Cin * Cin * 9 * 2);
  __bf16* w2p    = (__bf16*)alloc((size_t)Cout * Cin * 9 * 2);
  __bf16* wsp    = (__bf16*)alloc((size_t)Cout * Cin * 2);
  __bf16* y1     = (__bf16*)alloc((size_t)Nn * Cin * H * W * 2);
  __bf16* y1blur = (__bf16*)alloc((size_t)Nn * Cin * 65 * 65 * 2);
  __bf16* xblur  = (__bf16*)alloc((size_t)Nn * Cin * 63 * 63 * 2);

  // 1) spectral-norm scales
  sigma_kernel<<<1, 256, 0, stream>>>(w1_bar, u1, Cin, Cin * 9, vtmp, scales + 0);
  sigma_kernel<<<1, 256, 0, stream>>>(w2_bar, u2, Cout, Cin * 9, vtmp, scales + 1);
  sigma_kernel<<<1, 256, 0, stream>>>(ws_bar, us, Cout, Cin, vtmp, scales + 2);

  // 2) pack weights
  {
    long t1 = (long)Cin * Cin * 9;
    pack_weight<3, 3, 256><<<(int)((t1 + 255) / 256), 256, 0, stream>>>(w1_bar, scales + 0, w1p, Cin);
    long t2 = (long)Cout * Cin * 9;
    pack_weight<3, 3, 256><<<(int)((t2 + 255) / 256), 256, 0, stream>>>(w2_bar, scales + 1, w2p, Cout);
    long t3 = (long)Cout * Cin;
    pack_weight<1, 1, 256><<<(int)((t3 + 255) / 256), 256, 0, stream>>>(ws_bar, scales + 2, wsp, Cout);
  }

  // 3) x -> bf16
  {
    long nx = (long)Nn * Cin * H * W;
    cvt_bf16<<<(int)((nx + 255) / 256), 256, 0, stream>>>(x, xbf, nx);
  }

  // 4) conv1: 3x3 s1 p1, 256->256, bias+leaky, bf16 out
  {
    dim3 grid(Nn * H * (W / 32), 256 / 128);
    conv_wmma<0, 3, 3, 1, 1, 256, 256, 64, 64, 64, 64>
        <<<grid, 256, 0, stream>>>(xbf, w1p, b1, nullptr, y1);
  }

  // 5) blur(y1) pad=2 -> 65x65 bf16
  {
    long nb = (long)Nn * Cin * 65 * 65;
    blur_kernel<__bf16, 2, 64, 64, 65, 65>
        <<<(int)((nb + 255) / 256), 256, 0, stream>>>(y1, y1blur, Nn, Cin);
  }

  // 6) conv2: 3x3 s2 p0 on 65x65, 256->512, bias+leaky, /sqrt2 -> d_out
  {
    dim3 grid(Nn * 32, 512 / 128);
    conv_wmma<1, 3, 3, 2, 0, 256, 512, 65, 65, 32, 32>
        <<<grid, 256, 0, stream>>>(y1blur, w2p, b2, out, nullptr);
  }

  // 7) blur(x) pad=1 -> 63x63 bf16
  {
    long nb = (long)Nn * Cin * 63 * 63;
    blur_kernel<float, 1, 64, 64, 63, 63>
        <<<(int)((nb + 255) / 256), 256, 0, stream>>>(x, xblur, Nn, Cin);
  }

  // 8) skip: 1x1 s2 on 63x63, 256->512, accumulate /sqrt2 into d_out
  {
    dim3 grid(Nn * 32, 512 / 128);
    conv_wmma<2, 1, 1, 2, 0, 256, 512, 63, 63, 32, 32>
        <<<grid, 256, 0, stream>>>(xblur, wsp, nullptr, out, nullptr);
  }
}
